// E3NN_BasicConv_lengthless_54692113547898
// MI455X (gfx1250) — compile-verified, matrix operand order checked by reading, not run on hardware
//
#include <hip/hip_runtime.h>

typedef __attribute__((ext_vector_type(16))) _Float16 v16h;
typedef __attribute__((ext_vector_type(8)))  float    v8f;

#define WPB 4            // waves per block (wave32)
#define EPW 16           // edges per wave tile (WMMA N dim)
#define TPW 4            // tiles per wave (amortize weight staging)

// ---- normalization constants (path norms, Wigner norms, 1/sqrt(num_neighbors)) ----
constexpr float INV_SQRT10 = 0.31622776601683794f;
constexpr float INV_SQRT3  = 0.57735026918962576f;
constexpr float INV_SQRT2  = 0.70710678118654752f;
constexpr float INV_SQRT6  = 0.40824829046386302f;
constexpr float INV_SQRT5  = 0.44721359549995794f;
constexpr float NORM_S = 0.125f;                 // sqrt(1/64) for l=0 output paths
constexpr float NORM_V = 0.17677669529663687f;   // sqrt(3/96) for l=1 output paths
constexpr float C1 = NORM_S * INV_SQRT10;               // (0,0,0)
constexpr float C4 = NORM_S * INV_SQRT3 * INV_SQRT10;   // (1,1,0)
constexpr float C2 = NORM_V * INV_SQRT3 * INV_SQRT10;   // (0,1,1)
constexpr float C3 = NORM_V * INV_SQRT3 * INV_SQRT10;   // (1,0,1)
constexpr float C5 = NORM_V * INV_SQRT5 * INV_SQRT10;   // (1,2,1)

__global__ void e3nn_zero_kernel(float* __restrict__ out, int n) {
  int i = blockIdx.x * blockDim.x + threadIdx.x;
  if (i < n) out[i] = 0.0f;
}

// K index for 16-bit A/B WMMA fragments (ISA 7.12.2), simplified:
// element e of lane ln: K = (e<8 ? kb+e : 16+kb+(e-8)), kb = 8*(ln>=16)
__device__ __forceinline__ int wmma_kmap(int ln, int e) {
  int kb = (ln >> 4) << 3;
  return (e < 8) ? (kb + e) : (16 + kb + (e - 8));
}

__global__ __launch_bounds__(WPB * 32) void e3nn_conv_wmma_kernel(
    const float* __restrict__ x,  const int* __restrict__ ei,
    const float* __restrict__ ea,
    const float* __restrict__ w0, const float* __restrict__ w1,
    const float* __restrict__ w2, const float* __restrict__ w3,
    const float* __restrict__ w4,
    float* __restrict__ out, int E)
{
  // Weights pre-swizzled into A-fragment order: [path][Mtile][lane][16 halves]
  __shared__ __align__(32) _Float16 sW[5][2][32][16];
  // Per-wave per-edge staged u-vectors (f16), [wave][edge][u]
  __shared__ __align__(32) _Float16 sXS[WPB][EPW][32];
  __shared__ __align__(32) _Float16 sXV[3][WPB][EPW][32];
  __shared__ __align__(32) _Float16 sD [WPB][EPW][32];
  __shared__ __align__(32) _Float16 sZ [3][WPB][EPW][32];
  __shared__ float sS0[WPB][EPW];
  __shared__ float sS1[3][WPB][EPW];
  __shared__ int   sJ [WPB][EPW];

  const int tid  = threadIdx.x;
  const int wv   = tid >> 5;
  const int lane = tid & 31;

  // ---------- stage A-fragments of W^T for all 5 paths (once per block) ----------
  const float* Wp[5] = {w0, w1, w2, w3, w4};
  for (int idx = tid; idx < 5 * 2 * 32 * 16; idx += WPB * 32) {
    int e  = idx & 15;
    int ln = (idx >> 4) & 31;
    int t  = (idx >> 9) & 1;
    int p  = idx >> 10;
    int u  = wmma_kmap(ln, e);                   // K dim of A == mul index u
    int w  = (t << 4) + (ln & 15);               // M dim of A == output mul w
    sW[p][t][ln][e] = (_Float16)Wp[p][u * 32 + w];  // A[m][k] = W[u][w] (= W^T)
  }

  const long waveBase = ((long)blockIdx.x * WPB + wv) * (long)(EPW * TPW);
  const int  n   = lane & 15;                    // B fragment: N column = edge
  const int  kb  = (lane >> 4) << 3;
  const v16h* Wf = reinterpret_cast<const v16h*>(&sW[0][0][0][0]);

  for (int it = 0; it < TPW; ++it) {
    const long base = waveBase + (long)it * EPW;

    // ---------- stage per-edge data for this wave's 16-edge tile ----------
    {
      const int eLoc = lane >> 1, h = lane & 1;  // 2 lanes per edge, 16 u's each
      long e = base + eLoc;
      bool valid = e < (long)E;
      long ec = valid ? e : 0;
      int isrc = ei[ec];
      int jdst = ei[(long)E + ec];
      const float* er = ea + ec * 9;
      float s0  = valid ? er[0] : 0.f;           // l=0
      float s1a = valid ? er[1] : 0.f;           // l=1 (y,z,x order)
      float s1b = valid ? er[2] : 0.f;
      float s1c = valid ? er[3] : 0.f;
      float q0  = valid ? er[4] : 0.f;           // l=2: xy
      float q1  = valid ? er[5] : 0.f;           //      yz
      float q2  = valid ? er[6] : 0.f;           //      z^2
      float q3  = valid ? er[7] : 0.f;           //      zx
      float q4  = valid ? er[8] : 0.f;           //      x^2-y^2
      if (h == 0) {
        sS0[wv][eLoc] = s0;
        sS1[0][wv][eLoc] = s1a; sS1[1][wv][eLoc] = s1b; sS1[2][wv][eLoc] = s1c;
        sJ[wv][eLoc] = jdst;
      }
      // symmetric matrix S from l=2 SH == normalized w3j(1,2,1) contraction, axes (y,z,x)
      float Syy = -q2 * INV_SQRT6 - q4 * INV_SQRT2;
      float Szz =  2.f * q2 * INV_SQRT6;
      float Sxx = -q2 * INV_SQRT6 + q4 * INV_SQRT2;
      float Syz =  q1 * INV_SQRT2;
      float Szx =  q3 * INV_SQRT2;
      float Sxy =  q0 * INV_SQRT2;

      const float4* xr4 = reinterpret_cast<const float4*>(x + (long)isrc * 128);
      float xs[16], xvf[48];
      #pragma unroll
      for (int q = 0; q < 4; ++q) {              // l=0 block: channels [16h,16h+16)
        float4 f = xr4[4 * h + q];
        xs[4*q] = f.x; xs[4*q+1] = f.y; xs[4*q+2] = f.z; xs[4*q+3] = f.w;
      }
      #pragma unroll
      for (int q = 0; q < 12; ++q) {             // l=1 block: channels 32+48h ..
        float4 f = xr4[8 + 12 * h + q];
        xvf[4*q] = f.x; xvf[4*q+1] = f.y; xvf[4*q+2] = f.z; xvf[4*q+3] = f.w;
      }
      #pragma unroll
      for (int uu = 0; uu < 16; ++uu) {
        int u = 16 * h + uu;
        float vy = xvf[uu*3+0], vz = xvf[uu*3+1], vx = xvf[uu*3+2];
        float d  = vy * s1a + vz * s1b + vx * s1c;       // path (1,1,0) contraction
        float zy = Syy*vy + Syz*vz + Sxy*vx;             // path (1,2,1): S . xv
        float zz = Syz*vy + Szz*vz + Szx*vx;
        float zx = Sxy*vy + Szx*vz + Sxx*vx;
        sXS[wv][eLoc][u]    = (_Float16)xs[uu];
        sXV[0][wv][eLoc][u] = (_Float16)vy;
        sXV[1][wv][eLoc][u] = (_Float16)vz;
        sXV[2][wv][eLoc][u] = (_Float16)vx;
        sD [wv][eLoc][u]    = (_Float16)(d  * C4);
        sZ[0][wv][eLoc][u]  = (_Float16)(zy * C5);
        sZ[1][wv][eLoc][u]  = (_Float16)(zz * C5);
        sZ[2][wv][eLoc][u]  = (_Float16)(zx * C5);
      }
    }
    __syncthreads();

    // ---------- WMMA phase: D[w,n] = sum_u W[u,w] * col[u,n] over 16 edges ----------
    // B fragment = two contiguous 16-byte LDS chunks: halves [kb,kb+8) and [16+kb,16+kb+8)
    auto gatherB = [&](const _Float16* row) -> v16h {
      union { v16h v; float4 q[2]; } r;
      r.q[0] = *reinterpret_cast<const float4*>(row + kb);
      r.q[1] = *reinterpret_cast<const float4*>(row + 16 + kb);
      return r.v;
    };

    v16h fXS = gatherB(&sXS[wv][n][0]);
    v16h fXV[3], fZ[3];
    #pragma unroll
    for (int k = 0; k < 3; ++k) {
      fXV[k] = gatherB(&sXV[k][wv][n][0]);
      fZ[k]  = gatherB(&sZ [k][wv][n][0]);
    }
    v16h fD = gatherB(&sD[wv][n][0]);
    float s0 = sS0[wv][n];
    float s1v[3] = { sS1[0][wv][n], sS1[1][wv][n], sS1[2][wv][n] };

    #define LDW(p, t) (Wf[((p) * 2 + (t)) * 32 + lane])
    #define WMMA_ACC(A, B, C) \
      __builtin_amdgcn_wmma_f32_16x16x32_f16(false, (A), false, (B), (short)0, (C), false, false)

    v8f aS[2]    = { {0,0,0,0,0,0,0,0}, {0,0,0,0,0,0,0,0} };
    v8f aV[3][2] = { { {0,0,0,0,0,0,0,0}, {0,0,0,0,0,0,0,0} },
                    { {0,0,0,0,0,0,0,0}, {0,0,0,0,0,0,0,0} },
                    { {0,0,0,0,0,0,0,0}, {0,0,0,0,0,0,0,0} } };

    // scalar output: paths (0,0,0) and (1,1,0)
    {
      v16h cA = fXS * (_Float16)(s0 * C1);
      #pragma unroll
      for (int t = 0; t < 2; ++t) {
        aS[t] = WMMA_ACC(LDW(0, t), cA, aS[t]);
        aS[t] = WMMA_ACC(LDW(3, t), fD, aS[t]);
      }
    }
    // vector output: paths (0,1,1), (1,0,1), (1,2,1), per component k
    #pragma unroll
    for (int k = 0; k < 3; ++k) {
      v16h cB = fXS    * (_Float16)(s1v[k] * C2);
      v16h cC = fXV[k] * (_Float16)(s0     * C3);
      #pragma unroll
      for (int t = 0; t < 2; ++t) {
        aV[k][t] = WMMA_ACC(LDW(1, t), cB,    aV[k][t]);
        aV[k][t] = WMMA_ACC(LDW(2, t), cC,    aV[k][t]);
        aV[k][t] = WMMA_ACC(LDW(4, t), fZ[k], aV[k][t]);
      }
    }

    // ---------- scatter (C/D layout: M = r + 8*(lane>=16), N = lane&15) ----------
    int jrow = sJ[wv][n];
    float* op = out + (long)jrow * 128;
    const int mofs = (lane >> 4) * 8;
    #pragma unroll
    for (int t = 0; t < 2; ++t)
      #pragma unroll
      for (int r = 0; r < 8; ++r)
        __hip_atomic_fetch_add(&op[t * 16 + mofs + r], aS[t][r],
                               __ATOMIC_RELAXED, __HIP_MEMORY_SCOPE_AGENT);
    #pragma unroll
    for (int k = 0; k < 3; ++k)
      #pragma unroll
      for (int t = 0; t < 2; ++t)
        #pragma unroll
        for (int r = 0; r < 8; ++r)
          __hip_atomic_fetch_add(&op[32 + (t * 16 + mofs + r) * 3 + k], aV[k][t][r],
                                 __ATOMIC_RELAXED, __HIP_MEMORY_SCOPE_AGENT);

    __syncthreads();   // protect LDS slices before next iteration's staging
  }
}

extern "C" void kernel_launch(void* const* d_in, const int* in_sizes, int n_in,
                              void* d_out, int out_size, void* d_ws, size_t ws_size,
                              hipStream_t stream) {
  const float* x  = (const float*)d_in[0];
  const int*   ei = (const int*)d_in[1];
  const float* ea = (const float*)d_in[2];
  const float* w0 = (const float*)d_in[3];
  const float* w1 = (const float*)d_in[4];
  const float* w2 = (const float*)d_in[5];
  const float* w3 = (const float*)d_in[6];
  const float* w4 = (const float*)d_in[7];
  float* out = (float*)d_out;

  const int E = in_sizes[1] / 2;

  e3nn_zero_kernel<<<(out_size + 255) / 256, 256, 0, stream>>>(out, out_size);

  const int edgesPerBlock = WPB * TPW * EPW;     // 256
  const int blocks = (E + edgesPerBlock - 1) / edgesPerBlock;
  e3nn_conv_wmma_kernel<<<blocks, WPB * 32, 0, stream>>>(
      x, ei, ea, w0, w1, w2, w3, w4, out, E);
}
